// CausalSelfAttention_17557826306499
// MI455X (gfx1250) — matrix operand order repaired and checked by reading, not compiled, not run on hardware
//
#include <hip/hip_runtime.h>

// ---------------- problem constants (reference: B=2, T=2048, E=1024, H=16) --
#define BB 2
#define TT 2048
#define EE 1024
#define HH 16
#define HD 64
#define MM (BB * TT)          // 4096 rows
#define N3E (3 * EE)          // 3072

typedef __bf16 bf16;
typedef __attribute__((ext_vector_type(16))) __bf16 v16bf;
typedef __attribute__((ext_vector_type(8)))  __bf16 v8bf;
typedef __attribute__((ext_vector_type(8)))  float  v8f;

// D = A(16x32 bf16) * B(32x16 bf16) + C(16x16 f32)
__device__ __forceinline__ v8f wmma_bf16(v16bf a, v16bf b, v8f c) {
  return __builtin_amdgcn_wmma_f32_16x16x32_bf16(
      /*neg_a=*/false, a, /*neg_b=*/false, b,
      /*c_mod=*/(short)0, c, /*reuse_a=*/false, /*reuse_b=*/false);
}

// A-fragment (16x32, bf16 row-major source, row stride ld elements).
// ISA layout: lane<16 -> M=lane, elems 0..7 = K0..7,  8..15 = K16..23
//             lane>=16-> M=lane-16, elems 0..7 = K8..15, 8..15 = K24..31
__device__ __forceinline__ v16bf load_a_bf16(const bf16* base, int ld, int lane) {
  const int m = lane & 15, h = lane >> 4;
  const bf16* p = base + m * ld;
  v8bf lo = *(const v8bf*)(p + h * 8);
  v8bf hi = *(const v8bf*)(p + 16 + h * 8);
  v16bf r;
#pragma unroll
  for (int i = 0; i < 8; ++i) { r[i] = lo[i]; r[8 + i] = hi[i]; }
  return r;
}

// B-fragment (32x16) from a source stored "N-major" (src[n*ld + k]):
// lane holds column N = lane&15; elems 0..7 = K 8h..8h+7, 8..15 = K 16+8h.. (h=lane>>4).
__device__ __forceinline__ v16bf load_b_t(const bf16* base, int ld, int lane) {
  const bf16* p = base + (lane & 15) * ld + ((lane >> 4) << 3);
  v8bf lo = *(const v8bf*)(p);
  v8bf hi = *(const v8bf*)(p + 16);
  v16bf r;
#pragma unroll
  for (int i = 0; i < 8; ++i) { r[i] = lo[i]; r[8 + i] = hi[i]; }
  return r;
}

// gfx1250 async global -> LDS copy of 16 bytes (ASYNCcnt path).
// ISA 10.2 aperture rule: flat LDS address truncates to the LDS byte offset,
// so the low 32 bits of a __shared__ pointer are a valid DS address operand.
__device__ __forceinline__ void async_copy16(void* lds_dst, const void* gsrc) {
  unsigned int       ldsa = (unsigned int)(unsigned long long)lds_dst;
  unsigned long long ga   = (unsigned long long)gsrc;
  asm volatile("global_load_async_to_lds_b128 %0, %1, off"
               :: "v"(ldsa), "v"(ga) : "memory");
}
__device__ __forceinline__ void wait_async0() {
  asm volatile("s_wait_asynccnt 0" ::: "memory");
}
__device__ __forceinline__ void wait_async2() {   // allow 2 in flight
  asm volatile("s_wait_asynccnt 2" ::: "memory");
}
__device__ __forceinline__ void wait_ds0() {
  asm volatile("s_wait_dscnt 0" ::: "memory");
}

// ------------------------------------------------------------------ kernel 1
// Elementwise convert: wTq[n*E+k]=w_qkv[k,n]; wTp[n*E+k]=w_proj[k,n]; Xb=bf16(x)
__global__ void cvtw_kernel(const float* __restrict__ wq,
                            const float* __restrict__ wp,
                            const float* __restrict__ x,
                            bf16* __restrict__ wTq, bf16* __restrict__ wTp,
                            bf16* __restrict__ Xb) {
  const int idx = blockIdx.x * 256 + threadIdx.x;    // [0, 8M)
  const int NQ = N3E * EE;                           // 3M
  const int NP = EE * EE;                            // 1M
  if (idx < NQ) {
    const int n = idx / EE, k = idx - n * EE;
    wTq[idx] = (bf16)wq[k * N3E + n];
  } else if (idx < NQ + NP) {
    const int j = idx - NQ;
    const int n = j / EE, k = j - n * EE;
    wTp[j] = (bf16)wp[k * EE + n];
  } else {
    const int j = idx - NQ - NP;                     // [0, M*E)
    Xb[j] = (bf16)x[j];
  }
}

// ------------------------------------------------------------------ kernel 2
// qkv = x @ w_qkv + b ; each wave: 16x64 tile, software-pipelined K loop.
// Scatter into Q[B,H,T,64], K[B,H,T,64], Vt[B,H,64,T].
__global__ void qkv_kernel(const bf16* __restrict__ Xb,
                           const bf16* __restrict__ wTq,
                           const float* __restrict__ bias,
                           bf16* __restrict__ Qb, bf16* __restrict__ Kb,
                           bf16* __restrict__ Vt) {
  const int w = (blockIdx.x * blockDim.x + threadIdx.x) >> 5;
  const int lane = threadIdx.x & 31;
  const int MT = MM / 16;                  // 256
  const int mt = w % MT, nt4 = w / MT;     // nt4 in [0,48)
  const bf16* xrow = Xb + (mt * 16) * EE;
  const bf16* wrow = wTq + (nt4 * 64) * EE;

  v8f acc[4] = {};
  v16bf a = load_a_bf16(xrow, EE, lane);
  v16bf bfr[4];
#pragma unroll
  for (int g = 0; g < 4; ++g) bfr[g] = load_b_t(wrow + g * 16 * EE, EE, lane);

  for (int kt = 0; kt < EE / 32; ++kt) {
    v16bf an = a;
    v16bf bn[4];
    if (kt + 1 < EE / 32) {                    // prefetch next K-slice
      an = load_a_bf16(xrow + (kt + 1) * 32, EE, lane);
#pragma unroll
      for (int g = 0; g < 4; ++g)
        bn[g] = load_b_t(wrow + g * 16 * EE + (kt + 1) * 32, EE, lane);
    }
#pragma unroll
    for (int g = 0; g < 4; ++g) acc[g] = wmma_bf16(a, bfr[g], acc[g]);
    a = an;
#pragma unroll
    for (int g = 0; g < 4; ++g) bfr[g] = bn[g];
  }

  const int half = lane >> 4;
  const int col  = lane & 15;
  const int cat  = (nt4 * 64) >> 10;       // 0:Q 1:K 2:V (uniform per wave)
#pragma unroll
  for (int g = 0; g < 4; ++g) {
    const int n   = nt4 * 64 + g * 16 + col;
    const float bia = bias[n];
    const int np  = n & 1023;
    const int h   = np >> 6, d = np & 63;
#pragma unroll
    for (int r = 0; r < 8; ++r) {
      const int row = mt * 16 + half * 8 + r;     // global token row
      const int b_  = row >> 11;                  // /T
      const int t_  = row & (TT - 1);
      const bf16 bv = (bf16)(acc[g][r] + bia);
      if (cat == 0)      Qb[(((b_ * HH) + h) * TT + t_) * HD + d] = bv;
      else if (cat == 1) Kb[(((b_ * HH) + h) * TT + t_) * HD + d] = bv;
      else               Vt[(((b_ * HH) + h) * HD + d) * TT + t_] = bv;
    }
  }
}

// ------------------------------------------------------------------ kernel 3
// Flash attention. Block = 8 waves = 8 consecutive 16-query tiles of one
// (b,h). K/V 32-key tiles double-buffered in LDS via async copies, so tile
// kb+1 streams in while tile kb is consumed by WMMAs.
__global__ void attn_kernel(const bf16* __restrict__ Qb,
                            const bf16* __restrict__ Kb,
                            const bf16* __restrict__ Vt,
                            bf16* __restrict__ Y) {
  __shared__ __align__(32) bf16 kls[2][32 * HD];     // K tiles [key][d]   8 KB
  __shared__ __align__(32) bf16 vls[2][HD * 32];     // V tiles [d][key]   8 KB
  __shared__ __align__(32) bf16 pls[8][16 * 32];     // per-wave P staging 8 KB

  const int tid  = threadIdx.x;
  const int wid  = tid >> 5;
  const int lane = tid & 31;
  const int qgrp = blockIdx.x & 15;                  // 16 groups of 8 q-tiles
  const int h    = (blockIdx.x >> 4) & 15;
  const int b    = blockIdx.x >> 8;
  const int qt   = qgrp * 8 + wid;

  const bf16* qbase = Qb + (((b * HH) + h) * TT + qt * 16) * HD;
  const bf16* kbase = Kb + (((b * HH) + h) * TT) * HD;
  const bf16* vtb   = Vt + (((b * HH) + h) * HD) * TT;

  const v16bf qa0 = load_a_bf16(qbase,      HD, lane);   // d = 0..31
  const v16bf qa1 = load_a_bf16(qbase + 32, HD, lane);   // d = 32..63

  v8f o0 = {}, o1 = {}, o2 = {}, o3 = {};
  float mrow[8], lrow[8];
#pragma unroll
  for (int r = 0; r < 8; ++r) { mrow[r] = -1e30f; lrow[r] = 0.0f; }

  const int half  = lane >> 4;
  const int col   = lane & 15;
  const int qrow0 = qt * 16 + half * 8;
  const int kbmax = qt >> 1;                         // this wave's last block
  const int kbend = (qgrp * 8 + 7) >> 1;             // block-uniform bound
  bf16* pb = pls[wid];

  // this thread's chunk coordinates for cooperative staging
  const int kr = tid >> 3, kc = (tid & 7) * 8;       // K: 32 rows x 8 chunks
  const int vr = tid >> 2, vc = (tid & 3) * 8;       // V: 64 rows x 4 chunks

  // prologue: stage tile kb=0 into buffer 0 (2 async copies per thread)
  async_copy16(&kls[0][kr * HD + kc], kbase + kr * HD + kc);
  async_copy16(&vls[0][vr * 32 + vc], vtb + vr * TT + vc);

  for (int kb = 0; kb <= kbend; ++kb) {
    const int cur = kb & 1;
    if (kb < kbend) {                                // stream in next tile
      const int nxt = cur ^ 1;
      async_copy16(&kls[nxt][kr * HD + kc],
                   kbase + ((kb + 1) * 32 + kr) * HD + kc);
      async_copy16(&vls[nxt][vr * 32 + vc],
                   vtb + vr * TT + (kb + 1) * 32 + vc);
      wait_async2();                                 // oldest 2 (tile kb) done
    } else {
      wait_async0();
    }
    __syncthreads();                                 // tile kb visible to all

    if (kb <= kbmax) {
      const bf16* kt_ = kls[cur];
      const bf16* vt_ = vls[cur];
      // --- S = Q * K^T (two 16x16 tiles covering 32 keys) ----------------
      v8f s0 = {}, s1 = {};
      s0 = wmma_bf16(qa0, load_b_t(kt_,                HD, lane), s0);
      s0 = wmma_bf16(qa1, load_b_t(kt_ + 32,           HD, lane), s0);
      s1 = wmma_bf16(qa0, load_b_t(kt_ + 16 * HD,      HD, lane), s1);
      s1 = wmma_bf16(qa1, load_b_t(kt_ + 16 * HD + 32, HD, lane), s1);

      // --- causal mask + online softmax (f32) ----------------------------
      const int key0 = kb * 32 + col;
      const int key1 = key0 + 16;
      float alpha[8];
#pragma unroll
      for (int r = 0; r < 8; ++r) {
        const int q = qrow0 + r;
        float a0 = (key0 <= q) ? s0[r] * 0.125f : -1e30f;   // 1/sqrt(64)
        float a1 = (key1 <= q) ? s1[r] * 0.125f : -1e30f;
        float mx = fmaxf(a0, a1);
#pragma unroll
        for (int msk = 1; msk < 16; msk <<= 1)
          mx = fmaxf(mx, __shfl_xor(mx, msk, 32));
        const float nm = fmaxf(mrow[r], mx);
        alpha[r] = __expf(mrow[r] - nm);
        const float p0 = __expf(a0 - nm);
        const float p1 = __expf(a1 - nm);
        s0[r] = p0; s1[r] = p1;
        float rs = p0 + p1;
#pragma unroll
        for (int msk = 1; msk < 16; msk <<= 1)
          rs += __shfl_xor(rs, msk, 32);
        lrow[r] = lrow[r] * alpha[r] + rs;
        mrow[r] = nm;
      }
#pragma unroll
      for (int r = 0; r < 8; ++r) {
        o0[r] *= alpha[r]; o1[r] *= alpha[r];
        o2[r] *= alpha[r]; o3[r] *= alpha[r];
      }

      // --- C-layout P -> bf16 LDS -> A-layout fragment -------------------
#pragma unroll
      for (int r = 0; r < 8; ++r) {
        const int row = half * 8 + r;
        pb[row * 32 + col]      = (bf16)s0[r];
        pb[row * 32 + 16 + col] = (bf16)s1[r];
      }
      wait_ds0();                                    // intra-wave LDS RAW
      const v16bf pa = load_a_bf16(pb, 32, lane);

      // --- O += P * V ----------------------------------------------------
      o0 = wmma_bf16(pa, load_b_t(vt_ + 0 * 16 * 32, 32, lane), o0);
      o1 = wmma_bf16(pa, load_b_t(vt_ + 1 * 16 * 32, 32, lane), o1);
      o2 = wmma_bf16(pa, load_b_t(vt_ + 2 * 16 * 32, 32, lane), o2);
      o3 = wmma_bf16(pa, load_b_t(vt_ + 3 * 16 * 32, 32, lane), o3);
    }
    __syncthreads();           // all reads of tile kb done (frees buffer cur)
  }

  // --- normalize and write y[B,T,E] (bf16) -------------------------------
  bf16* yb = Y + (b * TT + qt * 16) * EE + h * HD;
#pragma unroll
  for (int r = 0; r < 8; ++r) {
    const float inv = 1.0f / lrow[r];
    bf16* yr = yb + (half * 8 + r) * EE;
    yr[0 * 16 + col] = (bf16)(o0[r] * inv);
    yr[1 * 16 + col] = (bf16)(o1[r] * inv);
    yr[2 * 16 + col] = (bf16)(o2[r] * inv);
    yr[3 * 16 + col] = (bf16)(o3[r] * inv);
  }
}

// ------------------------------------------------------------------ kernel 4
// out = y @ w_proj + b_proj (f32). Each wave: 16x64 tile, pipelined K loop.
__global__ void proj_kernel(const bf16* __restrict__ Y,
                            const bf16* __restrict__ wTp,
                            const float* __restrict__ bias,
                            float* __restrict__ out) {
  const int w = (blockIdx.x * blockDim.x + threadIdx.x) >> 5;
  const int lane = threadIdx.x & 31;
  const int MT = MM / 16;                  // 256
  const int mt = w % MT, nt4 = w / MT;     // nt4 in [0,16)
  const bf16* yrow = Y + (mt * 16) * EE;
  const bf16* wrow = wTp + (nt4 * 64) * EE;

  v8f acc[4] = {};
  v16bf a = load_a_bf16(yrow, EE, lane);
  v16bf bfr[4];
#pragma unroll
  for (int g = 0; g < 4; ++g) bfr[g] = load_b_t(wrow + g * 16 * EE, EE, lane);

  for (int kt = 0; kt < EE / 32; ++kt) {
    v16bf an = a;
    v16bf bn[4];
    if (kt + 1 < EE / 32) {
      an = load_a_bf16(yrow + (kt + 1) * 32, EE, lane);
#pragma unroll
      for (int g = 0; g < 4; ++g)
        bn[g] = load_b_t(wrow + g * 16 * EE + (kt + 1) * 32, EE, lane);
    }
#pragma unroll
    for (int g = 0; g < 4; ++g) acc[g] = wmma_bf16(a, bfr[g], acc[g]);
    a = an;
#pragma unroll
    for (int g = 0; g < 4; ++g) bfr[g] = bn[g];
  }

  const int half = lane >> 4;
  const int col  = lane & 15;
#pragma unroll
  for (int g = 0; g < 4; ++g) {
    const int n = nt4 * 64 + g * 16 + col;
    const float bi = bias[n];
#pragma unroll
    for (int r = 0; r < 8; ++r)
      out[(mt * 16 + half * 8 + r) * EE + n] = acc[g][r] + bi;
  }
}

// ---------------------------------------------------------------------------
extern "C" void kernel_launch(void* const* d_in, const int* in_sizes, int n_in,
                              void* d_out, int out_size, void* d_ws, size_t ws_size,
                              hipStream_t stream) {
  const float* x  = (const float*)d_in[0];
  const float* wq = (const float*)d_in[1];
  const float* bq = (const float*)d_in[2];
  const float* wp = (const float*)d_in[3];
  const float* bp = (const float*)d_in[4];
  float* out = (float*)d_out;

  char* ws = (char*)d_ws;
  const size_t SZ_WTQ = (size_t)N3E * EE * 2;          // 6 MB
  const size_t SZ_WTP = (size_t)EE * EE * 2;           // 2 MB
  const size_t SZ_X   = (size_t)MM * EE * 2;           // 8 MB
  const size_t SZ_QKV = (size_t)BB * HH * TT * HD * 2; // 8 MB each
  bf16* wTq = (bf16*)(ws);
  bf16* wTp = (bf16*)(ws + SZ_WTQ);
  bf16* Xb  = (bf16*)(ws + SZ_WTQ + SZ_WTP);
  bf16* Qb  = (bf16*)(ws + SZ_WTQ + SZ_WTP + SZ_X);
  bf16* Kb  = (bf16*)(ws + SZ_WTQ + SZ_WTP + SZ_X + SZ_QKV);
  bf16* Vt  = (bf16*)(ws + SZ_WTQ + SZ_WTP + SZ_X + 2 * SZ_QKV);
  bf16* Yb  = (bf16*)(ws + SZ_WTQ + SZ_WTP + SZ_X + 3 * SZ_QKV);

  // 1) convert weights (transposed) + x to bf16: 8M elements
  cvtw_kernel<<<(N3E * EE + EE * EE + MM * EE) / 256, 256, 0, stream>>>(
      wq, wp, x, wTq, wTp, Xb);
  // 2) QKV GEMM: 256 * 48 waves (16x64 tiles), 8 waves/block
  qkv_kernel<<<(256 * 48) / 8, 256, 0, stream>>>(Xb, wTq, bq, Qb, Kb, Vt);
  // 3) flash attention: 512 blocks of 8 q-tile waves, double-buffered LDS
  attn_kernel<<<512, 256, 0, stream>>>(Qb, Kb, Vt, Yb);
  // 4) output projection: 256 * 16 waves (16x64 tiles)
  proj_kernel<<<(256 * 16) / 8, 256, 0, stream>>>(Yb, wTp, bp, out);
}
